// T5MultiHeadAttention_44684839748198
// MI455X (gfx1250) — matrix operand order repaired and checked by reading, not compiled, not run on hardware
//
#include <hip/hip_runtime.h>
#include <stdint.h>

#define D_MODEL 768
#define N_HEADS 12
#define D_KH    64
#define MAX_REL 128
#define BB      2
#define SS      2048
#define MTOT    (BB*SS)          // 4096

typedef __bf16 bfx16 __attribute__((ext_vector_type(16)));
typedef float  fx8   __attribute__((ext_vector_type(8)));

union Frag {
    bfx16 v;
    uint4 q[2];
    unsigned short u[16];
};

__device__ __forceinline__ unsigned short f2bf(float f) {
    unsigned int u = __float_as_uint(f);
    u += 0x7FFFu + ((u >> 16) & 1u);      // round-to-nearest-even
    return (unsigned short)(u >> 16);
}

__device__ __forceinline__ fx8 wmma_bf16(const bfx16& a, const bfx16& b, const fx8& c) {
    return __builtin_amdgcn_wmma_f32_16x16x32_bf16(false, a, false, b, (short)0, c, false, false);
}

// Async DMA: each lane copies 16B global -> LDS; tracked by ASYNCcnt.
__device__ __forceinline__ void async_g2l_b128(uint32_t lds_off, uint64_t gaddr) {
    asm volatile("global_load_async_to_lds_b128 %0, %1, off"
                 :: "v"(lds_off), "v"(gaddr) : "memory");
}
__device__ __forceinline__ void wait_async0() {
    asm volatile("s_wait_asynccnt 0x0" ::: "memory");
}

// ---------------------------------------------------------------- converts
__global__ void cvt_f32_bf16(const float* __restrict__ in,
                             unsigned short* __restrict__ out, int n) {
    int i = blockIdx.x * 256 + threadIdx.x;
    if (i < n) out[i] = f2bf(in[i]);
}

// ---------------------------------------------------------------- GEMM
// C[m][n] = sum_k A[m][k] * Bm[n][k] + bias[n]
// Block tile: 128(M) x 64(N); 8 waves stacked in M share one async-staged,
// double-buffered 64x32 B tile in LDS.
// mode 0: write bf16 head-major [b,h,s,d]   (q / k projections)
// mode 1: write bf16 transposed [b,h,d,s]   (v projection)
// mode 2: write f32 plain [m][n]            (output projection)
__global__ __launch_bounds__(256)
void proj_gemm(const unsigned short* __restrict__ A,
               const unsigned short* __restrict__ Bm,
               const float* __restrict__ bias,
               unsigned short* __restrict__ outb,
               float* __restrict__ outf,
               int mode)
{
    __shared__ __align__(16) unsigned short btile[2][64 * 32];   // 2 x 4 KB

    const int tid  = threadIdx.x;
    const int wid  = tid >> 5;
    const int lane = tid & 31;
    const int hl   = lane >> 4;
    const int lr   = lane & 15;

    const int ntN = D_MODEL / 64;              // 12
    const int tm  = blockIdx.x / ntN;          // 0..31
    const int tn  = blockIdx.x % ntN;
    const int m0  = tm * 128 + wid * 16;
    const int n0  = tn * 64;

    // per-thread DMA assignment: 64 rows x 4 chunks of 8 bf16 (16B)
    const int srow   = tid >> 2;               // 0..63
    const int schunk = tid & 3;                // 0..3
    const unsigned short* gsrc =
        Bm + (size_t)(n0 + srow) * D_MODEL + schunk * 8;
    const uint32_t ldst0 =
        (uint32_t)(uintptr_t)(&btile[0][srow * 32 + schunk * 8]);
    const uint32_t ldst1 =
        (uint32_t)(uintptr_t)(&btile[1][srow * 32 + schunk * 8]);

    fx8 acc[4];
    for (int i = 0; i < 4; ++i)
        for (int j = 0; j < 8; ++j) acc[i][j] = 0.f;

    const unsigned short* arow = A + (size_t)(m0 + lr) * D_MODEL;

    // prologue: stage k=0 tile into buffer 0
    async_g2l_b128(ldst0, (uint64_t)(uintptr_t)gsrc);

    for (int k = 0; k < D_MODEL; k += 32) {
        const int cur = (k >> 5) & 1;
        wait_async0();          // this wave's staged chunk landed
        __syncthreads();        // all waves' chunks landed; prev readers done

        if (k + 32 < D_MODEL)   // prefetch next tile into the other buffer
            async_g2l_b128(cur ? ldst0 : ldst1,
                           (uint64_t)(uintptr_t)(gsrc + k + 32));

        Frag a;
        a.q[0] = *(const uint4*)(arow + k + hl * 8);
        a.q[1] = *(const uint4*)(arow + k + 16 + hl * 8);
#pragma unroll
        for (int jn = 0; jn < 4; ++jn) {
            const unsigned short* brow = &btile[cur][(jn * 16 + lr) * 32 + hl * 16];
            Frag b;
            b.q[0] = *(const uint4*)(brow - hl * 8);
            b.q[1] = *(const uint4*)(brow - hl * 8 + 16);
            acc[jn] = wmma_bf16(a.v, b.v, acc[jn]);
        }
    }

#pragma unroll
    for (int jn = 0; jn < 4; ++jn) {
#pragma unroll
        for (int r = 0; r < 8; ++r) {
            int m = m0 + r + hl * 8;
            int n = n0 + jn * 16 + lr;
            float v = acc[jn][r] + bias[n];
            int b = m >> 11, s = m & (SS - 1);
            int h = n >> 6,  d = n & 63;
            if (mode == 0) {
                outb[(((size_t)(b * N_HEADS + h) * SS + s) << 6) + d] = f2bf(v);
            } else if (mode == 1) {
                outb[((size_t)(b * N_HEADS + h) * D_KH + d) * SS + s] = f2bf(v);
            } else {
                outf[(size_t)m * D_MODEL + n] = v;
            }
        }
    }
}

// ---------------------------------------------------------------- fused attention
// One block = 16 query rows of one (b,h).  8 waves, 256 threads.
// LDS: probs[16][2048] f32 (128 KB) + part[8][16][16] f32 (8 KB).
__global__ __launch_bounds__(256)
void attention_kernel(const unsigned short* __restrict__ qh,   // [b,h,s,d] bf16
                      const unsigned short* __restrict__ kh,   // [b,h,s,d] bf16
                      const unsigned short* __restrict__ vT,   // [b,h,d,s] bf16
                      const int*            __restrict__ mask, // [b,1,S,S]
                      const float*          __restrict__ rel,  // [257]
                      unsigned short*       __restrict__ ctx)  // [b*S, 768] bf16
{
    extern __shared__ float smem[];
    float* probs = smem;            // 16*2048
    float* part  = smem + 16 * SS;  // 8*16*16

    const int tid  = threadIdx.x;
    const int wid  = tid >> 5;
    const int lane = tid & 31;
    const int hl   = lane >> 4;
    const int lr   = lane & 15;

    const int tilesPerBH = SS / 16;              // 128
    const int bh = blockIdx.x / tilesPerBH;
    const int qb = (blockIdx.x % tilesPerBH) * 16;
    const int b  = bh / N_HEADS;
    const int h  = bh % N_HEADS;

    // A fragments for the 16 query rows (K = 0..63 split into two 32-chunks)
    const unsigned short* qrow = qh + ((size_t)bh * SS + qb + lr) * D_KH;
    Frag a0, a1;
    a0.q[0] = *(const uint4*)(qrow + 0  + hl * 8);
    a0.q[1] = *(const uint4*)(qrow + 16 + hl * 8);
    a1.q[0] = *(const uint4*)(qrow + 32 + hl * 8);
    a1.q[1] = *(const uint4*)(qrow + 48 + hl * 8);

    // ---- phase 1: scores = q @ k^T * 1/8 + relbias, masked, -> LDS
    for (int j2 = 0; j2 < 4; ++j2) {
        int n0 = wid * 256 + j2 * 64;
        fx8 acc[4];
        for (int i = 0; i < 4; ++i)
            for (int j = 0; j < 8; ++j) acc[i][j] = 0.f;
#pragma unroll
        for (int jn = 0; jn < 4; ++jn) {
            const unsigned short* krow =
                kh + ((size_t)bh * SS + n0 + jn * 16 + lr) * D_KH;
            Frag bf;
            bf.q[0] = *(const uint4*)(krow + hl * 8);
            bf.q[1] = *(const uint4*)(krow + 16 + hl * 8);
            acc[jn] = wmma_bf16(a0.v, bf.v, acc[jn]);
            bf.q[0] = *(const uint4*)(krow + 32 + hl * 8);
            bf.q[1] = *(const uint4*)(krow + 48 + hl * 8);
            acc[jn] = wmma_bf16(a1.v, bf.v, acc[jn]);
        }
#pragma unroll
        for (int jn = 0; jn < 4; ++jn) {
#pragma unroll
            for (int r = 0; r < 8; ++r) {
                int mrow = r + hl * 8;          // 0..15
                int mq   = qb + mrow;
                int n    = n0 + jn * 16 + lr;
                float sc = acc[jn][r] * 0.125f;
                int dist = n - mq;
                dist = dist < -MAX_REL ? -MAX_REL : (dist > MAX_REL ? MAX_REL : dist);
                sc += rel[dist + MAX_REL];
                if (mask[((size_t)b * SS + mq) * SS + n] == 0) sc = -1e9f;
                probs[mrow * SS + n] = sc;
            }
        }
    }
    __syncthreads();

    // ---- phase 2: row softmax (wave handles 2 rows, lanes stride 32)
    for (int rr = 0; rr < 2; ++rr) {
        int row = wid * 2 + rr;
        float* p = probs + row * SS;
        float mx = -3.4e38f;
        for (int n = lane; n < SS; n += 32) mx = fmaxf(mx, p[n]);
#pragma unroll
        for (int o = 16; o; o >>= 1) mx = fmaxf(mx, __shfl_xor(mx, o, 32));
        float sum = 0.f;
        for (int n = lane; n < SS; n += 32) {
            float e = __expf(p[n] - mx);
            p[n] = e;
            sum += e;
        }
#pragma unroll
        for (int o = 16; o; o >>= 1) sum += __shfl_xor(sum, o, 32);
        float inv = 1.0f / sum;
        for (int n = lane; n < SS; n += 32) p[n] *= inv;
    }
    __syncthreads();

    // ---- phase 3: out = probs @ v  (K=2048 split: 2 waves per 16-col tile)
    int ntile = wid & 3;     // which 16 columns of d_k
    int khalf = wid >> 2;    // which half of K
    int n0 = ntile * 16;
    const unsigned short* vrow = vT + ((size_t)bh * D_KH + n0 + lr) * SS;
    fx8 acc;
    for (int j = 0; j < 8; ++j) acc[j] = 0.f;
    for (int ks = khalf * 1024; ks < khalf * 1024 + 1024; ks += 32) {
        Frag a;
#pragma unroll
        for (int i = 0; i < 16; ++i) {
            int k = ks + (i & 7) + ((i >> 3) << 4) + hl * 8;
            a.u[i] = f2bf(probs[lr * SS + k]);
        }
        Frag bf;
        bf.q[0] = *(const uint4*)(vrow + ks + hl * 8);
        bf.q[1] = *(const uint4*)(vrow + ks + 16 + hl * 8);
        acc = wmma_bf16(a.v, bf.v, acc);
    }
#pragma unroll
    for (int r = 0; r < 8; ++r)
        part[wid * 256 + (r + hl * 8) * 16 + lr] = acc[r];
    __syncthreads();

    for (int idx = tid; idx < 1024; idx += 256) {
        int m = idx >> 6, n = idx & 63;
        int nt = n >> 4, nn = n & 15;
        float v = part[nt * 256 + m * 16 + nn] + part[(nt + 4) * 256 + m * 16 + nn];
        ctx[((size_t)(b * SS) + qb + m) * D_MODEL + h * D_KH + n] = f2bf(v);
    }
}

// ---------------------------------------------------------------- host
extern "C" void kernel_launch(void* const* d_in, const int* in_sizes, int n_in,
                              void* d_out, int out_size, void* d_ws, size_t ws_size,
                              hipStream_t stream) {
    const float* Q    = (const float*)d_in[0];
    const float* K    = (const float*)d_in[1];
    const float* V    = (const float*)d_in[2];
    const int*   mask = (const int*)  d_in[3];
    const float* Wq   = (const float*)d_in[4];
    const float* bq   = (const float*)d_in[5];
    const float* Wk   = (const float*)d_in[6];
    const float* bk   = (const float*)d_in[7];
    const float* Wv   = (const float*)d_in[8];
    const float* bv   = (const float*)d_in[9];
    const float* Wo   = (const float*)d_in[10];
    const float* bo   = (const float*)d_in[11];
    const float* rel  = (const float*)d_in[12];
    float* out = (float*)d_out;

    const size_t WB = (size_t)D_MODEL * D_MODEL;   // 589824
    const size_t XB = (size_t)MTOT * D_MODEL;      // 3145728

    uintptr_t p = (uintptr_t)d_ws;
    auto take = [&](size_t elems) {
        unsigned short* r = (unsigned short*)p;
        p += ((elems * 2 + 255) & ~(size_t)255);
        return r;
    };
    unsigned short* Wqb = take(WB);
    unsigned short* Wkb = take(WB);
    unsigned short* Wvb = take(WB);
    unsigned short* Wob = take(WB);
    unsigned short* Xq  = take(XB);
    unsigned short* Xk  = take(XB);
    unsigned short* Xv  = take(XB);
    unsigned short* qh  = take(XB);   // [b,h,s,d]
    unsigned short* kh  = take(XB);   // [b,h,s,d]
    unsigned short* vT  = take(XB);   // [b,h,d,s]
    unsigned short* ctx = take(XB);   // [b*s, 768]

    auto cvt = [&](const float* src, unsigned short* dst, size_t n) {
        cvt_f32_bf16<<<dim3((unsigned)((n + 255) / 256)), dim3(256), 0, stream>>>(src, dst, (int)n);
    };
    cvt(Wq, Wqb, WB);
    cvt(Wk, Wkb, WB);
    cvt(Wv, Wvb, WB);
    cvt(Wo, Wob, WB);
    cvt(Q,  Xq,  XB);
    cvt(K,  Xk,  XB);
    cvt(V,  Xv,  XB);

    // projections: 32 x 12 block tiles of 128x64
    proj_gemm<<<dim3(384), dim3(256), 0, stream>>>(Xq, Wqb, bq, qh, nullptr, 0);
    proj_gemm<<<dim3(384), dim3(256), 0, stream>>>(Xk, Wkb, bk, kh, nullptr, 0);
    proj_gemm<<<dim3(384), dim3(256), 0, stream>>>(Xv, Wvb, bv, vT, nullptr, 1);

    const size_t smem = (16 * (size_t)SS + 8 * 16 * 16) * sizeof(float); // 139264 B
    (void)hipFuncSetAttribute((const void*)attention_kernel,
                              hipFuncAttributeMaxDynamicSharedMemorySize, (int)smem);
    attention_kernel<<<dim3(BB * N_HEADS * (SS / 16)), dim3(256), smem, stream>>>(
        qh, kh, vT, mask, rel, ctx);

    proj_gemm<<<dim3(384), dim3(256), 0, stream>>>(ctx, Wob, bo, nullptr, out, 2);
}
